// MyMutilHeadAttention_62371515072629
// MI455X (gfx1250) — matrix operand order repaired and checked by reading, not compiled, not run on hardware
//
#include <hip/hip_runtime.h>
#include <hip/hip_bf16.h>

// Problem constants (N=4, S=1024, D=1024, H=16, HD=64)
#define S_  1024
#define D_  1024
#define NB_ 4
#define H_  16
#define HD_ 64

typedef __attribute__((ext_vector_type(16))) __bf16 v16bf;
typedef __attribute__((ext_vector_type(8)))  __bf16 v8bf;
typedef __attribute__((ext_vector_type(8)))  float  v8f;
typedef __attribute__((ext_vector_type(8)))  float  v8fv;

// Native fp32 -> bf16
static __device__ __forceinline__ __bf16 f2bf(float f) { return (__bf16)f; }

static __device__ __forceinline__ v8f wmma_bf16(v16bf a, v16bf b, v8f c) {
  // D = A(16x32 bf16) * B(32x16 bf16) + C(16x16 f32)
  return __builtin_amdgcn_wmma_f32_16x16x32_bf16(false, a, false, b, (short)0, c,
                                                 false, false);
}

// A fragment (16x32 bf16). Caller passes A already offset to (row0, k0).
static __device__ __forceinline__ v16bf load_a_bf(const __bf16* A, int lda, int lane) {
  int m = lane & 15, kh = lane >> 4;
  const __bf16* p = A + (size_t)m * lda + kh * 16;
  v8bf lo = *(const v8bf*)p;
  v8bf hi = *(const v8bf*)(p + 8);
  v16bf a;
#pragma unroll
  for (int i = 0; i < 8; ++i) { a[i] = lo[i]; a[8 + i] = hi[i]; }
  return a;
}

// B fragment from Bt = B^T stored N x K row-major (16 contiguous bf16 per lane).
static __device__ __forceinline__ v16bf load_bt_bf(const __bf16* Bt, int ldb, int lane) {
  int n = lane & 15, kh = lane >> 4;
  return *(const v16bf*)(Bt + (size_t)n * ldb + kh * 16);
}

// C/D 16x16 f32 layout: col = lane&15, row = (lane>>4)*8 + v.
static __device__ __forceinline__ void store_c_bf(__bf16* C, int ldc, v8f c, int lane) {
  int n = lane & 15, mh = lane >> 4;
#pragma unroll
  for (int v = 0; v < 8; ++v) C[(size_t)(mh * 8 + v) * ldc + n] = f2bf(c[v]);
}

// ---------------------------------------------------------------------------
// Kernel A: bulk fp32 -> bf16 conversion (8 elems / thread)
// ---------------------------------------------------------------------------
__global__ void cvt_f32_bf16_kernel(const float* __restrict__ src,
                                    __bf16* __restrict__ dst) {
  int i = blockIdx.x * blockDim.x + threadIdx.x;
  v8fv x = *(const v8fv*)(src + (size_t)i * 8);
  v8bf y;
#pragma unroll
  for (int j = 0; j < 8; ++j) y[j] = f2bf(x[j]);
  *(v8bf*)(dst + (size_t)i * 8) = y;
}

// ---------------------------------------------------------------------------
// Kernel 0: mask (int32 [S,S]) -> maskT (bf16 [S,S], transposed)
// ---------------------------------------------------------------------------
__global__ void mask_transpose_kernel(const int* __restrict__ mask,
                                      __bf16* __restrict__ maskT) {
  int idx = blockIdx.x * blockDim.x + threadIdx.x;
  int i = idx >> 10, j = idx & (S_ - 1);
  maskT[(size_t)j * S_ + i] = f2bf((float)mask[(size_t)i * S_ + j]);
}

// ---------------------------------------------------------------------------
// Kernel 1: QKV projections (all-bf16). X[4096,1024] @ W^T + b.
// z=0 -> Qh [N,H,S,64]; z=1 -> Kt [N,H,64,S] (transposed!); z=2 -> Vt [N,H,64,S].
// Per-wave 32x64 C tile (2x4 WMMA). 2048 tiles.
// ---------------------------------------------------------------------------
__global__ __launch_bounds__(256) void qkv_proj_kernel(
    const __bf16* __restrict__ X,
    const __bf16* __restrict__ Wqb, const float* __restrict__ bq,
    const __bf16* __restrict__ Wkb, const float* __restrict__ bk,
    const __bf16* __restrict__ Wvb, const float* __restrict__ bv,
    __bf16* __restrict__ Qh, __bf16* __restrict__ Kt, __bf16* __restrict__ Vt) {
  int lane = threadIdx.x & 31;
  int tile = blockIdx.x * 8 + (threadIdx.x >> 5);      // 2048 tiles (128 x 16)
  int tn = tile & 15, tm = tile >> 4;
  int r0 = tm * 32, c0 = tn * 64;
  int mode = blockIdx.z;
  const __bf16* W = (mode == 0) ? Wqb : (mode == 1) ? Wkb : Wvb;
  const float* bias = (mode == 0) ? bq : (mode == 1) ? bk : bv;

  v8f acc[2][4] = {};
  for (int k0 = 0; k0 < D_; k0 += 32) {
    v16bf a0 = load_a_bf(X + (size_t)r0 * D_ + k0, D_, lane);
    v16bf a1 = load_a_bf(X + (size_t)(r0 + 16) * D_ + k0, D_, lane);
#pragma unroll
    for (int ni = 0; ni < 4; ++ni) {
      v16bf b = load_bt_bf(W + (size_t)(c0 + ni * 16) * D_ + k0, D_, lane);
      acc[0][ni] = wmma_bf16(a0, b, acc[0][ni]);
      acc[1][ni] = wmma_bf16(a1, b, acc[1][ni]);
    }
  }

  int n_ = lane & 15, mh = lane >> 4;
#pragma unroll
  for (int mi = 0; mi < 2; ++mi)
#pragma unroll
    for (int ni = 0; ni < 4; ++ni)
#pragma unroll
      for (int v = 0; v < 8; ++v) {
        int r = r0 + mi * 16 + mh * 8 + v;             // [0,4096)
        int c = c0 + ni * 16 + n_;                     // [0,1024)
        __bf16 val = f2bf(acc[mi][ni][v] + bias[c]);
        int b = r >> 10, s = r & (S_ - 1);
        int h = c >> 6,  d = c & (HD_ - 1);
        size_t bh = (size_t)(b * H_ + h);
        if (mode == 0)      Qh[(bh * S_ + s) * HD_ + d] = val;   // [S,64]
        else if (mode == 1) Kt[(bh * HD_ + d) * S_ + s] = val;   // [64,S]
        else                Vt[(bh * HD_ + d) * S_ + s] = val;   // [64,S]
      }
}

// ---------------------------------------------------------------------------
// Kernel 2: KtM = K^T @ mask per head.  M=64, N=1024(j), K=1024(s).
// (QK^T)M == Q(K^T M): turns the 1024^3 mask GEMM into two skinny GEMMs.
// A = Kt rows (contiguous), Bt = maskT rows (contiguous).
// Output stored TRANSPOSED: KtMt [N,H,S(j),64(d)] so it is the Bt operand of
// the next GEMM with contiguous fragment loads.
// ---------------------------------------------------------------------------
__global__ __launch_bounds__(256) void ktm_kernel(
    const __bf16* __restrict__ Kt, const __bf16* __restrict__ maskT,
    __bf16* __restrict__ KtMt) {
  int lane = threadIdx.x & 31;
  int tile = blockIdx.x * 8 + (threadIdx.x >> 5);      // 32 tiles (2 x 16)
  int tn = tile & 15, tm = tile >> 4;
  int bh = blockIdx.y;
  const __bf16* A = Kt + (size_t)bh * HD_ * S_;        // [64,S], ld=S
  __bf16* outT = KtMt + (size_t)bh * S_ * HD_;         // [S,64]
  int r0 = tm * 32, c0 = tn * 64;                      // r0: d, c0: j

  v8f acc[2][4] = {};
  for (int k0 = 0; k0 < S_; k0 += 32) {
    v16bf a0 = load_a_bf(A + (size_t)r0 * S_ + k0, S_, lane);
    v16bf a1 = load_a_bf(A + (size_t)(r0 + 16) * S_ + k0, S_, lane);
#pragma unroll
    for (int ni = 0; ni < 4; ++ni) {
      v16bf b = load_bt_bf(maskT + (size_t)(c0 + ni * 16) * S_ + k0, S_, lane);
      acc[0][ni] = wmma_bf16(a0, b, acc[0][ni]);
      acc[1][ni] = wmma_bf16(a1, b, acc[1][ni]);
    }
  }
  int n_ = lane & 15, mh = lane >> 4;
#pragma unroll
  for (int mi = 0; mi < 2; ++mi)
#pragma unroll
    for (int ni = 0; ni < 4; ++ni)
#pragma unroll
      for (int v = 0; v < 8; ++v) {
        int d = r0 + mi * 16 + mh * 8 + v;             // [0,64)
        int j = c0 + ni * 16 + n_;                     // [0,1024)
        outT[(size_t)j * HD_ + d] = f2bf(acc[mi][ni][v]);
      }
}

// ---------------------------------------------------------------------------
// Kernel 3: masked = Q @ KtM (K=64, 2 k-steps) ; attn = softmax(masked/8).
// Block owns a 16-row strip; wave w covers cols [w*128, w*128+128).
// Softmax in registers via shfl_xor (wave32 halves) + small LDS cross-wave.
// ---------------------------------------------------------------------------
__global__ __launch_bounds__(256) void masked_softmax_kernel(
    const __bf16* __restrict__ Qh, const __bf16* __restrict__ KtMt,
    __bf16* __restrict__ attn) {
  int lane = threadIdx.x & 31;
  int w = threadIdx.x >> 5;                             // 0..7
  int r0 = blockIdx.x * 16;
  int bh = blockIdx.y;
  const __bf16* Q  = Qh   + (size_t)bh * S_ * HD_;      // [S,64]
  const __bf16* Bt = KtMt + (size_t)bh * S_ * HD_;      // [S,64] (= (KtM)^T)
  __bf16* rowbase = attn + (size_t)bh * S_ * S_ + (size_t)r0 * S_;

  __shared__ float red[8 * 16];
  __shared__ float rstat[16];

  v8f acc[8] = {};
#pragma unroll
  for (int k0 = 0; k0 < HD_; k0 += 32) {
    v16bf a = load_a_bf(Q + (size_t)r0 * HD_ + k0, HD_, lane);
#pragma unroll
    for (int t = 0; t < 8; ++t) {
      v16bf b = load_bt_bf(Bt + (size_t)(w * 128 + t * 16) * HD_ + k0, HD_, lane);
      acc[t] = wmma_bf16(a, b, acc[t]);                 // A reused 8x per k-step
    }
  }
#pragma unroll
  for (int t = 0; t < 8; ++t)
#pragma unroll
    for (int v = 0; v < 8; ++v) acc[t][v] *= 0.125f;    // 1/sqrt(64)

  int mh = lane >> 4, n_ = lane & 15;

  // ---- row max ----
  float pm[8];
#pragma unroll
  for (int v = 0; v < 8; ++v) {
    float m = acc[0][v];
#pragma unroll
    for (int t = 1; t < 8; ++t) m = fmaxf(m, acc[t][v]);
    for (int s = 1; s <= 8; s <<= 1) m = fmaxf(m, __shfl_xor(m, s, 32));
    pm[v] = m;
  }
  if (n_ == 0) {
#pragma unroll
    for (int v = 0; v < 8; ++v) red[w * 16 + mh * 8 + v] = pm[v];
  }
  __syncthreads();
  if (threadIdx.x < 16) {
    float m = red[threadIdx.x];
    for (int ww = 1; ww < 8; ++ww) m = fmaxf(m, red[ww * 16 + threadIdx.x]);
    rstat[threadIdx.x] = m;
  }
  __syncthreads();
  float rowmax[8];
#pragma unroll
  for (int v = 0; v < 8; ++v) rowmax[v] = rstat[mh * 8 + v];
  __syncthreads();

  // ---- exp + row sum ----
  float ps[8];
#pragma unroll
  for (int v = 0; v < 8; ++v) {
    float sum = 0.f;
#pragma unroll
    for (int t = 0; t < 8; ++t) {
      float e = __expf(acc[t][v] - rowmax[v]);
      acc[t][v] = e;
      sum += e;
    }
    for (int s = 1; s <= 8; s <<= 1) sum += __shfl_xor(sum, s, 32);
    ps[v] = sum;
  }
  if (n_ == 0) {
#pragma unroll
    for (int v = 0; v < 8; ++v) red[w * 16 + mh * 8 + v] = ps[v];
  }
  __syncthreads();
  if (threadIdx.x < 16) {
    float ssum = 0.f;
    for (int ww = 0; ww < 8; ++ww) ssum += red[ww * 16 + threadIdx.x];
    rstat[threadIdx.x] = ssum;
  }
  __syncthreads();

  // ---- normalize + store attn bf16 ----
#pragma unroll
  for (int v = 0; v < 8; ++v) {
    float inv = 1.0f / rstat[mh * 8 + v];
    int row = mh * 8 + v;
#pragma unroll
    for (int t = 0; t < 8; ++t)
      rowbase[(size_t)row * S_ + w * 128 + t * 16 + n_] = f2bf(acc[t][v] * inv);
  }
}

// ---------------------------------------------------------------------------
// Kernel 4: head_out = attn @ V per (n,h). M=1024, N=64, K=1024.
// One 32x64 tile per wave covers the full head width.
// ---------------------------------------------------------------------------
__global__ __launch_bounds__(256) void attn_v_kernel(
    const __bf16* __restrict__ attn, const __bf16* __restrict__ Vt,
    __bf16* __restrict__ head_out) {
  int lane = threadIdx.x & 31;
  int tm = blockIdx.x * 8 + (threadIdx.x >> 5);         // 32 row tiles
  int bh = blockIdx.y;
  int b = bh >> 4, h = bh & 15;
  const __bf16* A = attn + (size_t)bh * S_ * S_;
  const __bf16* B = Vt + (size_t)bh * HD_ * S_;          // Bt = V^T rows
  __bf16* out = head_out + (size_t)b * S_ * D_ + h * HD_;
  int r0 = tm * 32;

  v8f acc[2][4] = {};
  for (int k0 = 0; k0 < S_; k0 += 32) {
    v16bf a0 = load_a_bf(A + (size_t)r0 * S_ + k0, S_, lane);
    v16bf a1 = load_a_bf(A + (size_t)(r0 + 16) * S_ + k0, S_, lane);
#pragma unroll
    for (int ni = 0; ni < 4; ++ni) {
      v16bf bb = load_bt_bf(B + (size_t)(ni * 16) * S_ + k0, S_, lane);
      acc[0][ni] = wmma_bf16(a0, bb, acc[0][ni]);
      acc[1][ni] = wmma_bf16(a1, bb, acc[1][ni]);
    }
  }
#pragma unroll
  for (int mi = 0; mi < 2; ++mi)
#pragma unroll
    for (int ni = 0; ni < 4; ++ni)
      store_c_bf(out + (size_t)(r0 + mi * 16) * D_ + ni * 16, D_, acc[mi][ni], lane);
}

// ---------------------------------------------------------------------------
// Kernel 5: out = head_out @ Wo^T + bo.  M=4096, N=1024, K=1024. fp32 out.
// ---------------------------------------------------------------------------
__global__ __launch_bounds__(256) void out_proj_kernel(
    const __bf16* __restrict__ Xh, const __bf16* __restrict__ Wob,
    const float* __restrict__ bo, float* __restrict__ out) {
  int lane = threadIdx.x & 31;
  int tile = blockIdx.x * 8 + (threadIdx.x >> 5);       // 2048 tiles (128 x 16)
  int tn = tile & 15, tm = tile >> 4;
  int r0 = tm * 32, c0 = tn * 64;

  v8f acc[2][4] = {};
  for (int k0 = 0; k0 < D_; k0 += 32) {
    v16bf a0 = load_a_bf(Xh + (size_t)r0 * D_ + k0, D_, lane);
    v16bf a1 = load_a_bf(Xh + (size_t)(r0 + 16) * D_ + k0, D_, lane);
#pragma unroll
    for (int ni = 0; ni < 4; ++ni) {
      v16bf b = load_bt_bf(Wob + (size_t)(c0 + ni * 16) * D_ + k0, D_, lane);
      acc[0][ni] = wmma_bf16(a0, b, acc[0][ni]);
      acc[1][ni] = wmma_bf16(a1, b, acc[1][ni]);
    }
  }
  int n_ = lane & 15, mh = lane >> 4;
#pragma unroll
  for (int mi = 0; mi < 2; ++mi)
#pragma unroll
    for (int ni = 0; ni < 4; ++ni)
#pragma unroll
      for (int v = 0; v < 8; ++v) {
        int c = c0 + ni * 16 + n_;
        out[(size_t)(r0 + mi * 16 + mh * 8 + v) * D_ + c] = acc[mi][ni][v] + bo[c];
      }
}

// ---------------------------------------------------------------------------
extern "C" void kernel_launch(void* const* d_in, const int* in_sizes, int n_in,
                              void* d_out, int out_size, void* d_ws, size_t ws_size,
                              hipStream_t stream) {
  (void)in_sizes; (void)n_in; (void)out_size; (void)ws_size;

  const float* query = (const float*)d_in[0];
  // d_in[1] key_t, d_in[2] value: unused (reference computes Q,K,V from `query`)
  const int*   mask  = (const int*)d_in[3];
  const float* Wq = (const float*)d_in[4];  const float* bq = (const float*)d_in[5];
  const float* Wk = (const float*)d_in[6];  const float* bk = (const float*)d_in[7];
  const float* Wv = (const float*)d_in[8];  const float* bv = (const float*)d_in[9];
  const float* Wo = (const float*)d_in[10]; const float* bo = (const float*)d_in[11];
  float* out = (float*)d_out;

  char* ws = (char*)d_ws;
  __bf16* Qh    = (__bf16*)(ws + ((size_t)0  << 20));   //  8 MB [N,H,S,64]
  __bf16* Kt    = (__bf16*)(ws + ((size_t)8  << 20));   //  8 MB [N,H,64,S]
  __bf16* Vt    = (__bf16*)(ws + ((size_t)16 << 20));   //  8 MB [N,H,64,S]
  __bf16* Ho    = (__bf16*)(ws + ((size_t)24 << 20));   //  8 MB [N,S,D]
  __bf16* maskT = (__bf16*)(ws + ((size_t)32 << 20));   //  2 MB [S,S]
  __bf16* Xb    = (__bf16*)(ws + ((size_t)34 << 20));   //  8 MB query bf16
  __bf16* Wqb   = (__bf16*)(ws + ((size_t)42 << 20));   //  2 MB
  __bf16* Wkb   = (__bf16*)(ws + ((size_t)44 << 20));   //  2 MB
  __bf16* Wvb   = (__bf16*)(ws + ((size_t)46 << 20));   //  2 MB
  __bf16* Wob   = (__bf16*)(ws + ((size_t)48 << 20));   //  2 MB
  __bf16* KtMt  = (__bf16*)(ws + ((size_t)50 << 20));   //  8 MB [N,H,S,64]
  __bf16* attn  = (__bf16*)(ws + ((size_t)58 << 20));   // 128 MB [N,H,S,S]

  // fp32 -> bf16 staging (once)
  cvt_f32_bf16_kernel<<<dim3(2048), dim3(256), 0, stream>>>(query, Xb); // 4M elems
  cvt_f32_bf16_kernel<<<dim3(512),  dim3(256), 0, stream>>>(Wq, Wqb);   // 1M elems
  cvt_f32_bf16_kernel<<<dim3(512),  dim3(256), 0, stream>>>(Wk, Wkb);
  cvt_f32_bf16_kernel<<<dim3(512),  dim3(256), 0, stream>>>(Wv, Wvb);
  cvt_f32_bf16_kernel<<<dim3(512),  dim3(256), 0, stream>>>(Wo, Wob);
  mask_transpose_kernel<<<dim3(4096), dim3(256), 0, stream>>>(mask, maskT);

  qkv_proj_kernel<<<dim3(256, 1, 3), dim3(256), 0, stream>>>(
      Xb, Wqb, bq, Wkb, bk, Wvb, bv, Qh, Kt, Vt);
  ktm_kernel<<<dim3(4, 64), dim3(256), 0, stream>>>(Kt, maskT, KtMt);
  masked_softmax_kernel<<<dim3(64, 64), dim3(256), 0, stream>>>(Qh, KtMt, attn);
  attn_v_kernel<<<dim3(4, 64), dim3(256), 0, stream>>>(attn, Vt, Ho);
  out_proj_kernel<<<dim3(256), dim3(256), 0, stream>>>(Ho, Wob, bo, out);
}
